// Decoder_spirals_82231443849262
// MI455X (gfx1250) — compile-verified, matrix operand order checked by reading, not run on hardware
//
#include <hip/hip_runtime.h>
#include <hip/hip_bf16.h>
#include <type_traits>

// CDNA5 / gfx1250 wave32 WMMA GEMM pipeline (Neural3DMM decoder).
// - LDS tiles stored directly in v_wmma_f32_16x16x32_bf16 fragment layout
//   (each lane loads its 32B fragment contiguously via ds_load_b128).
// - Double-buffered LDS software pipeline: next tile's global loads are in
//   flight while WMMAs consume the current buffer; one barrier per K-step.
// - Guard-free main loop; single guarded tail tile for K % 32 != 0.

typedef __bf16 bf16_t;
typedef __attribute__((ext_vector_type(16))) __bf16 v16bf;
typedef __attribute__((ext_vector_type(2)))  __bf16 v2bf;
typedef __attribute__((ext_vector_type(8)))  float  v8f;

#define BM 64
#define BK 32
#define NTHREADS 128   // 4 waves of 32

// Fragment K-pair base for pair-index p (0..7) and k-half h (0..1):
//   K = ((p<4)? 2p : 16+2(p-4)) + 8h ; the pair covers K and K+1
__device__ __forceinline__ int pairK(int p, int h) {
    return ((p < 4) ? (2 * p) : (16 + 2 * (p - 4))) + h * 8;
}

// C = act(A @ B + bias)
// MODE 0: A is row-major (M x K) fp32 (A shared across batch when aBatch=0)
// MODE 1: A rows gathered through spiral table S: row m = b*Vg+v,
//         col kk -> (k=kk>>Fshift, fi=kk&mask), elem = X[(b*Vg+S[v*Kspir+k])<<Fshift | fi]
// NT: number of 16-wide N tiles per block (block tile = 64 x 16*NT)
// NGUARD: true only when N is not a multiple of BN (conv0, N=3)
template <int MODE, int NT, bool NGUARD>
__global__ __launch_bounds__(NTHREADS)
void wmma_gemm_dec(const float* __restrict__ A,
                   const float* __restrict__ Bm,
                   const float* __restrict__ bias,   // nullable
                   float* __restrict__ C,
                   int M, int N, int K,
                   long aBatch, long bBatch, long cBatch,
                   const int* __restrict__ S, int Vg, int Kspir, int Fshift,
                   int act, int zeroLast)
{
    constexpr int BN = NT * 16;
    constexpr int NSHIFT = (BN == 128) ? 7 : 6;

    // Double-buffered fragment-layout LDS: [buf][tile][lane][slot]
    __shared__ __align__(32) bf16_t lA[2][4][32][16];
    __shared__ __align__(32) bf16_t lB[2][NT][32][16];
    constexpr int ABYTES = 4 * 32 * 16 * (int)sizeof(bf16_t);
    constexpr int BBYTES = NT * 32 * 16 * (int)sizeof(bf16_t);

    const int tid  = threadIdx.x;
    const int wave = tid >> 5;
    const int lane = tid & 31;
    const int half = lane >> 4;
    const int ln15 = lane & 15;

    const int bm = blockIdx.y * BM;
    const int bn = blockIdx.x * BN;
    const int bz = blockIdx.z;

    const float* Ab = A  + (long)bz * aBatch;
    const float* Bb = Bm + (long)bz * bBatch;
    float*       Cb = C  + (long)bz * cBatch;

    // ---- A staging constants (pair id q=tid&15 -> k-half/pair; rows (tid>>4)+8e) ----
    const int qa  = tid & 15;
    const int ha  = qa >> 3;
    const int pa  = qa & 7;
    const int pkA = pairK(pa, ha);
    const int mB4 = tid >> 4;
    const int Fmask = (1 << Fshift) - 1;

    long       aOff[8];      // MODE0: row*K + pkA ; MODE1: batch-base element offset
    const int* sRow[8];
    v2bf*      aSt[8];       // LDS store slots (buffer 0)
    #pragma unroll
    for (int e = 0; e < 8; ++e) {
        int m  = mB4 + 8 * e;
        int gm = bm + m;
        if (gm > M - 1) gm = M - 1;          // clamp: stores guarded in epilogue
        if (MODE == 0) {
            aOff[e] = (long)gm * K + pkA;
            sRow[e] = nullptr;
        } else {
            int bb = gm / Vg;
            aOff[e] = ((long)bb * Vg) << Fshift;
            sRow[e] = S + (gm - bb * Vg) * Kspir;
        }
        aSt[e] = (v2bf*)&lA[0][m >> 4][(m & 15) + 16 * ha][2 * pa];
    }

    // ---- B staging constants ----
    const int nB  = tid & (BN - 1);
    const int gnB = bn + nB;
    const int jnB = NGUARD ? ((gnB > N - 1) ? (N - 1) : gnB) : gnB;
    int   pkB[NT * 2];
    long  bOff[NT * 2];      // pkB*N + jnB (element offset within K-slab)
    v2bf* bSt[NT * 2];       // LDS store slots (buffer 0)
    #pragma unroll
    for (int e = 0; e < NT * 2; ++e) {
        int q  = (tid + e * NTHREADS) >> NSHIFT;
        pkB[e] = pairK(q & 7, q >> 3);
        bOff[e] = (long)pkB[e] * N + jnB;
        bSt[e] = (v2bf*)&lB[0][nB >> 4][(nB & 15) + 16 * (q >> 3)][2 * (q & 7)];
    }

    v8f acc[NT] = {};

    // staging registers (live across the WMMA chain in the pipelined loop)
    float a0[8], a1[8];
    float c0[NT * 2], c1[NT * 2];

    // ---- pipeline stages ----
    auto loadTile = [&](int k0, auto KGc) {
        constexpr bool KG = decltype(KGc)::value;
        const int gkA = k0 + pkA;
        if (MODE == 0) {
            const float* A0 = Ab + k0;
            #pragma unroll
            for (int e = 0; e < 8; ++e) {
                if (KG) {
                    long r = aOff[e] - pkA;          // row*K
                    int i0 = gkA;     if (i0 > K - 1) i0 = K - 1;
                    int i1 = gkA + 1; if (i1 > K - 1) i1 = K - 1;
                    float l0 = Ab[r + i0];
                    float l1 = Ab[r + i1];
                    a0[e] = (gkA     < K) ? l0 : 0.0f;
                    a1[e] = (gkA + 1 < K) ? l1 : 0.0f;
                } else {
                    a0[e] = A0[aOff[e]];
                    a1[e] = A0[aOff[e] + 1];
                }
            }
        } else {
            const int kA  = gkA >> Fshift;
            const int fiA = gkA & Fmask;
            #pragma unroll
            for (int e = 0; e < 8; ++e) {
                int src = sRow[e][kA];
                float2 v = *(const float2*)(Ab + aOff[e] + (((long)src) << Fshift) + fiA);
                a0[e] = v.x; a1[e] = v.y;
            }
        }
        const float* B0 = Bb + (long)k0 * N;
        #pragma unroll
        for (int e = 0; e < NT * 2; ++e) {
            if (KG) {
                int kb = k0 + pkB[e];
                int i0 = kb;     if (i0 > K - 1) i0 = K - 1;
                int i1 = kb + 1; if (i1 > K - 1) i1 = K - 1;
                float l0 = Bb[(long)i0 * N + jnB];
                float l1 = Bb[(long)i1 * N + jnB];
                bool okn = !NGUARD || (gnB < N);
                c0[e] = ((kb     < K) && okn) ? l0 : 0.0f;
                c1[e] = ((kb + 1 < K) && okn) ? l1 : 0.0f;
            } else {
                float l0 = B0[bOff[e]];
                float l1 = B0[bOff[e] + N];
                if (NGUARD) {
                    bool okn = gnB < N;
                    l0 = okn ? l0 : 0.0f;
                    l1 = okn ? l1 : 0.0f;
                }
                c0[e] = l0; c1[e] = l1;
            }
        }
    };

    auto storeTile = [&](int buf) {
        const int off = buf ? 1 : 0;
        #pragma unroll
        for (int e = 0; e < 8; ++e) {
            v2bf pk; pk.x = (bf16_t)a0[e]; pk.y = (bf16_t)a1[e];
            *(v2bf*)((char*)aSt[e] + off * ABYTES) = pk;
        }
        #pragma unroll
        for (int e = 0; e < NT * 2; ++e) {
            v2bf pk; pk.x = (bf16_t)c0[e]; pk.y = (bf16_t)c1[e];
            *(v2bf*)((char*)bSt[e] + off * BBYTES) = pk;
        }
    };

    auto compute = [&](int buf) {
        v16bf af = *(const v16bf*)&lA[buf][wave][lane][0];
        v16bf bfr[NT];
        #pragma unroll
        for (int nt = 0; nt < NT; ++nt)
            bfr[nt] = *(const v16bf*)&lB[buf][nt][lane][0];
        #pragma unroll
        for (int nt = 0; nt < NT; ++nt)
            acc[nt] = __builtin_amdgcn_wmma_f32_16x16x32_bf16(
                false, af, false, bfr[nt], (short)0, acc[nt], false, false);
    };

    // ---- software-pipelined K loop (one barrier per step) ----
    const int Kmain = K & ~(BK - 1);   // >= BK for all layers here
    loadTile(0, std::integral_constant<bool, false>{});
    storeTile(0);
    __syncthreads();
    int cur = 0;
    for (int k0 = BK; k0 < Kmain; k0 += BK) {
        loadTile(k0, std::integral_constant<bool, false>{});  // loads in flight...
        compute(cur);                                          // ...during WMMA
        storeTile(cur ^ 1);
        __syncthreads();
        cur ^= 1;
    }
    if (Kmain < K) {
        loadTile(Kmain, std::integral_constant<bool, true>{});
        compute(cur);
        storeTile(cur ^ 1);
        __syncthreads();
        cur ^= 1;
    }
    compute(cur);

    // ---- fused epilogue: bias + ELU + dummy-vertex zeroing ----
    const int mo = wave * 16;
    #pragma unroll
    for (int nt = 0; nt < NT; ++nt) {
        int gn = bn + nt * 16 + ln15;
        if (NGUARD && gn >= N) continue;
        float bv = bias ? bias[gn] : 0.0f;
        #pragma unroll
        for (int r = 0; r < 8; ++r) {
            int gm = bm + mo + r + half * 8;
            if (gm >= M) continue;
            float v = acc[nt][r] + bv;
            if (act) v = (v > 0.0f) ? v : (__expf(v) - 1.0f);
            if (zeroLast) {
                int bb = gm / Vg;
                if (gm - bb * Vg == Vg - 1) v = 0.0f;
            }
            Cb[(long)gm * N + gn] = v;
        }
    }
}

extern "C" void kernel_launch(void* const* d_in, const int* in_sizes, int n_in,
                              void* d_out, int out_size, void* d_ws, size_t ws_size,
                              hipStream_t stream) {
    (void)in_sizes; (void)n_in; (void)out_size; (void)ws_size;

    const float* z   = (const float*)d_in[0];
    const float* U0  = (const float*)d_in[1];
    const float* U1  = (const float*)d_in[2];
    const float* U2  = (const float*)d_in[3];
    const int*   S0  = (const int*)d_in[4];
    const int*   S1  = (const int*)d_in[5];
    const int*   S2  = (const int*)d_in[6];
    const float* Wfc = (const float*)d_in[7];
    const float* bfc = (const float*)d_in[8];
    const float* W0  = (const float*)d_in[9];
    const float* b0  = (const float*)d_in[10];
    const float* W1  = (const float*)d_in[11];
    const float* b1  = (const float*)d_in[12];
    const float* W2  = (const float*)d_in[13];
    const float* b2  = (const float*)d_in[14];
    float* out = (float*)d_out;
    float* ws  = (float*)d_ws;

    const int B = 16;
    const int V0 = 11794, V1 = 2949, V2 = 738, V3 = 185;
    // fp32 activation ping-pong:
    //  regA: xfc -> x2c -> x1c   (cap 3,019,776 floats)
    //  regB: x2u -> x1u -> x0u   (cap 12,077,056 floats)
    float* regA = ws;
    float* regB = ws + 3019776;

    dim3 blk(NTHREADS);

    // 1) fc: (16x128)@(128x23680)+bfc -> regA (B,185,128)
    wmma_gemm_dec<0, 8, false><<<dim3(23680 / 128, 1, 1), blk, 0, stream>>>(
        z, Wfc, bfc, regA, 16, 23680, 128, 0, 0, 0,
        nullptr, 0, 0, 0, 0, 0);

    // 2) up2: U2 (738x185) @ x[b](185x128) -> regB, batched over z
    wmma_gemm_dec<0, 8, false><<<dim3(1, (V2 + BM - 1) / BM, B), blk, 0, stream>>>(
        U2, regA, nullptr, regB, V2, 128, V3,
        0, (long)V3 * 128, (long)V2 * 128,
        nullptr, 0, 0, 0, 0, 0);

    // 3) conv2: gather(S2,K=12,F=128) @ W0(1536x128)+b0, ELU -> regA
    wmma_gemm_dec<1, 8, false><<<dim3(1, (B * V2 + BM - 1) / BM, 1), blk, 0, stream>>>(
        regB, W0, b0, regA, B * V2, 128, 12 * 128, 0, 0, 0,
        S2, V2, 12, 7, 1, 1);

    // 4) up1: U1 (2949x738) @ x[b](738x128) -> regB
    wmma_gemm_dec<0, 8, false><<<dim3(1, (V1 + BM - 1) / BM, B), blk, 0, stream>>>(
        U1, regA, nullptr, regB, V1, 128, V2,
        0, (long)V2 * 128, (long)V1 * 128,
        nullptr, 0, 0, 0, 0, 0);

    // 5) conv1: gather(S1,K=14,F=128) @ W1(1792x64)+b1, ELU -> regA
    wmma_gemm_dec<1, 4, false><<<dim3(1, (B * V1 + BM - 1) / BM, 1), blk, 0, stream>>>(
        regB, W1, b1, regA, B * V1, 64, 14 * 128, 0, 0, 0,
        S1, V1, 14, 7, 1, 1);

    // 6) up0: U0 (11794x2949) @ x[b](2949x64) -> regB  (U0 reused from L2 across batch)
    wmma_gemm_dec<0, 4, false><<<dim3(1, (V0 + BM - 1) / BM, B), blk, 0, stream>>>(
        U0, regA, nullptr, regB, V0, 64, V1,
        0, (long)V1 * 64, (long)V0 * 64,
        nullptr, 0, 0, 0, 0, 0);

    // 7) conv0: gather(S0,K=20,F=64) @ W2(1280x3)+b2, identity -> d_out
    wmma_gemm_dec<1, 4, true><<<dim3(1, (B * V0 + BM - 1) / BM, 1), blk, 0, stream>>>(
        regB, W2, b2, out, B * V0, 3, 20 * 64, 0, 0, 0,
        S0, V0, 20, 6, 0, 1);
}
